// TABGNNFused_73770358276769
// MI455X (gfx1250) — matrix operand order repaired and checked by reading, not compiled, not run on hardware
//
#include <hip/hip_runtime.h>
#include <stdint.h>

// ---------------------------------------------------------------------------
// Problem constants (from the reference):
//   N=50000 nodes, E=200000 edges, B=16384 target edges
//   C=64, NH=128, HEADS=8 (dh=8), S=6, FF=64, ED=448, FD=320
// ---------------------------------------------------------------------------
#define N_NODES 50000
#define N_EDGES 200000
#define N_B     16384
#define TF_CHUNK 20000   // sequences per transformer chunk; 7*TF_CHUNK % 16 == 0

typedef __attribute__((ext_vector_type(16))) __bf16        v16bf;
typedef __attribute__((ext_vector_type(8)))  float         v8f;
typedef __attribute__((ext_vector_type(8)))  unsigned int  v8u;

__device__ __forceinline__ uint16_t f32_to_bf16(float f) {
  uint32_t u = __builtin_bit_cast(uint32_t, f);
  uint32_t r = u + 0x7FFFu + ((u >> 16) & 1u);   // round-to-nearest-even
  return (uint16_t)(r >> 16);
}
// order-preserving float<->uint map for atomic max/min on f32
__device__ __forceinline__ unsigned f2ord(float f) {
  unsigned u = __builtin_bit_cast(unsigned, f);
  return (u & 0x80000000u) ? ~u : (u | 0x80000000u);
}
__device__ __forceinline__ float ord2f(unsigned u) {
  unsigned v = (u & 0x80000000u) ? (u & 0x7FFFFFFFu) : ~u;
  return __builtin_bit_cast(float, v);
}

// ---------------------------------------------------------------------------
// WMMA GEMM: act( A[M,K](bf16) @ W[N,K](bf16)^T + bias[N] ) -> Cf (f32) / Cb (bf16)
// act: 0 = none, 1 = relu, 2 = leaky_relu(0.01). Cf/Cb individually nullable.
// Requirements (satisfied by this model): K%32==0, N%64==0, M%16==0.
// Block = 256 threads = 8 waves; wave w -> rows [blockIdx.y*128 + 16w, +16),
// cols [blockIdx.x*64, +64) = 4 accumulators of v_wmma_f32_16x16x32_bf16.
// Inner loop: 2x b128 (A frag) + 4 x 2x b128 (B frags) + 4 wmma, no convert VALU.
//
// Fragment layouts used here (contiguous k order, same for A and B):
//   A (16x32 bf16): lane&15 = M row; lane>>4 selects K {0..15} vs {16..31} (32B contig)
//   B (32x16 bf16): lane&15 = N col; lane>>4 selects K {0..15} vs {16..31} (32B contig)
//   C/D (16x16 f32): col = lane&15, row = vgpr + 8*(lane>>4)
// ---------------------------------------------------------------------------
__global__ __launch_bounds__(256)
void gemm_bf16(const uint16_t* __restrict__ A, const uint16_t* __restrict__ W,
               const float* __restrict__ bias, float* __restrict__ Cf,
               uint16_t* __restrict__ Cb, int M, int N, int K, int act)
{
  const int lane = threadIdx.x & 31;
  const int wv   = threadIdx.x >> 5;
  const int m0   = blockIdx.y * 128 + wv * 16;
  if (m0 >= M) return;                       // wave-uniform (M % 16 == 0)
  const int n0   = blockIdx.x * 64;
  const int r    = lane & 15;
  const int kh   = lane >> 4;

  v8f acc[4];
  #pragma unroll
  for (int j = 0; j < 4; ++j)
    #pragma unroll
    for (int g = 0; g < 8; ++g) acc[j][g] = 0.0f;

  const uint16_t* arow = A + (size_t)(m0 + r) * K;
  for (int k0 = 0; k0 < K; k0 += 32) {
    v8u au;
    const uint32_t* pa0 = (const uint32_t*)(arow + k0 + kh * 16);       // K = k0+kh*16+[0..7]
    const uint32_t* pa1 = (const uint32_t*)(arow + k0 + 8 + kh * 16);   // +8
    #pragma unroll
    for (int i2 = 0; i2 < 4; ++i2) { au[i2] = pa0[i2]; au[4 + i2] = pa1[i2]; }
    v16bf af = __builtin_bit_cast(v16bf, au);

    #pragma unroll
    for (int j = 0; j < 4; ++j) {
      const uint32_t* q = (const uint32_t*)(W + (size_t)(n0 + j*16 + r) * K + k0 + kh * 16);
      v8u bu;
      #pragma unroll
      for (int i2 = 0; i2 < 8; ++i2) bu[i2] = q[i2];
      v16bf bfr = __builtin_bit_cast(v16bf, bu);
      acc[j] = __builtin_amdgcn_wmma_f32_16x16x32_bf16(false, af, false, bfr,
                                                       (short)0, acc[j], false, false);
    }
  }

  #pragma unroll
  for (int j = 0; j < 4; ++j) {
    const int col = n0 + j * 16 + r;
    const float bv = bias[col];
    #pragma unroll
    for (int g = 0; g < 8; ++g) {
      const int row = m0 + g + (kh ? 8 : 0);
      float v = acc[j][g] + bv;
      if (act == 1)      v = fmaxf(v, 0.f);
      else if (act == 2) v = v > 0.f ? v : 0.01f * v;
      size_t o = (size_t)row * N + col;
      if (Cf) Cf[o] = v;
      if (Cb) Cb[o] = f32_to_bf16(v);
    }
  }
}

// ---------------------------------------------------------------------------
// Elementwise / reduction kernels
// ---------------------------------------------------------------------------
__global__ void cast_bf16_kernel(const float* __restrict__ s, uint16_t* __restrict__ d, int n) {
  int i = blockIdx.x * blockDim.x + threadIdx.x;
  if (i < n) d[i] = f32_to_bf16(s[i]);
}

// x (N,8) @ node_emb.w(128,8)^T + b -> out (N,128) f32
__global__ void node_emb_kernel(const float* __restrict__ x, const float* __restrict__ w,
                                const float* __restrict__ b, float* __restrict__ out, long long n)
{
  long long i = (long long)blockIdx.x * blockDim.x + threadIdx.x;
  if (i >= n) return;
  int c = (int)(i & 127); long long row = i >> 7;
  const float* xr = x + row * 8;
  const float* wr = w + c * 8;
  float s = b[c];
  #pragma unroll
  for (int k = 0; k < 8; ++k) s += xr[k] * wr[k];
  out[i] = s;
}

// tokens (nseq,7,64): token 0 = cls, tokens 1..6 = attr[seq, t-1, :]; f32 + bf16 mirror
__global__ void build_tok_kernel(const float* __restrict__ cls, const float* __restrict__ attr,
                                 float* __restrict__ tok, uint16_t* __restrict__ tokb,
                                 long long n)
{
  long long i = (long long)blockIdx.x * blockDim.x + threadIdx.x;
  if (i >= n) return;
  int c = (int)(i % 448); long long b = i / 448;
  int t = c >> 6, ch = c & 63;
  float v = (t == 0) ? cls[ch] : attr[(b * 6 + (t - 1)) * 64 + ch];
  tok[i] = v;
  tokb[i] = f32_to_bf16(v);
}

// 8-head attention on 7 tokens; qkv (nseq,7,[3][8][8]) f32; o (nseq,7,64) bf16
__global__ __launch_bounds__(64)
void attn7_kernel(const float* __restrict__ qkv, uint16_t* __restrict__ o)
{
  __shared__ float sh[7 * 192];
  const size_t base = (size_t)blockIdx.x * (7 * 192);
  for (int i = threadIdx.x; i < 7 * 192; i += 64) sh[i] = qkv[base + i];
  __syncthreads();
  int t = threadIdx.x;
  if (t >= 56) return;
  int h = t / 7, qi = t % 7;
  float q[8];
  #pragma unroll
  for (int d = 0; d < 8; ++d) q[d] = sh[qi * 192 + h * 8 + d];
  float s[7], mx = -3.4e38f;
  #pragma unroll
  for (int j = 0; j < 7; ++j) {
    float a = 0.f;
    #pragma unroll
    for (int d = 0; d < 8; ++d) a += q[d] * sh[j * 192 + 64 + h * 8 + d];
    s[j] = a * 0.35355339059327373f;   // 8^-0.5
    mx = fmaxf(mx, s[j]);
  }
  float se = 0.f;
  #pragma unroll
  for (int j = 0; j < 7; ++j) { s[j] = expf(s[j] - mx); se += s[j]; }
  float inv = 1.f / se;
  uint16_t* orow = o + (size_t)blockIdx.x * 448 + (size_t)qi * 64 + h * 8;
  #pragma unroll
  for (int d = 0; d < 8; ++d) {
    float a = 0.f;
    #pragma unroll
    for (int j = 0; j < 7; ++j) a += s[j] * sh[j * 192 + 128 + h * 8 + d];
    orow[d] = f32_to_bf16(a * inv);
  }
}

// y = LayerNorm(x + r) row-wise; r nullable; optional bf16 mirror yb.
// One wave per row, block = 8 rows.
__global__ __launch_bounds__(256)
void add_ln_kernel(const float* __restrict__ x, const float* __restrict__ r,
                   const float* __restrict__ w, const float* __restrict__ b,
                   float* __restrict__ y, uint16_t* __restrict__ yb, int rows, int dim)
{
  int row = blockIdx.x * 8 + (threadIdx.x >> 5);
  if (row >= rows) return;
  int lane = threadIdx.x & 31;
  const float* xr = x + (size_t)row * dim;
  const float* rr = r ? r + (size_t)row * dim : nullptr;
  float s = 0.f, s2 = 0.f;
  for (int d = lane; d < dim; d += 32) {
    float v = xr[d] + (rr ? rr[d] : 0.f);
    s += v; s2 += v * v;
  }
  #pragma unroll
  for (int off = 16; off; off >>= 1) { s += __shfl_xor(s, off, 32); s2 += __shfl_xor(s2, off, 32); }
  float m = s / dim;
  float inv = rsqrtf(s2 / dim - m * m + 1e-5f);
  float* yr = y + (size_t)row * dim;
  uint16_t* ybr = yb ? yb + (size_t)row * dim : nullptr;
  for (int d = lane; d < dim; d += 32) {
    float v = xr[d] + (rr ? rr[d] : 0.f);
    float o = (v - m) * inv * w[d] + b[d];
    yr[d] = o;
    if (ybr) ybr[d] = f32_to_bf16(o);
  }
}

// out (E,384) bf16 = [ xg[i0[e]], xg[i1[e]], e3[e] ]
__global__ void cat3_kernel(const float* __restrict__ xg, const int* __restrict__ i0,
                            const int* __restrict__ i1, const float* __restrict__ e3,
                            uint16_t* __restrict__ out, long long n)
{
  long long i = (long long)blockIdx.x * blockDim.x + threadIdx.x;
  if (i >= n) return;
  int c = (int)(i % 384); long long e = i / 384;
  float v;
  if (c < 128)       v = xg[(size_t)i0[e] * 128 + c];
  else if (c < 256)  v = xg[(size_t)i1[e] * 128 + (c - 128)];
  else               v = e3[e * 128 + (c - 256)];
  out[i] = f32_to_bf16(v);
}

__global__ void pna_scatter_kernel(const float* __restrict__ m, const int* __restrict__ dst,
                                   float* __restrict__ sum, float* __restrict__ sumsq,
                                   unsigned* __restrict__ mx, unsigned* __restrict__ mn,
                                   long long n)
{
  long long i = (long long)blockIdx.x * blockDim.x + threadIdx.x;
  if (i >= n) return;
  int c = (int)(i & 127); long long e = i >> 7;
  size_t d = (size_t)dst[e] * 128 + c;
  float v = m[i];
  atomicAdd(&sum[d], v);
  atomicAdd(&sumsq[d], v * v);
  unsigned o = f2ord(v);
  atomicMax(&mx[d], o);
  atomicMin(&mn[d], o);
}

__global__ void count_kernel(const int* __restrict__ idx, float* __restrict__ cnt, int n) {
  int i = blockIdx.x * blockDim.x + threadIdx.x;
  if (i < n) atomicAdd(&cnt[idx[i]], 1.0f);
}

// sum_n log(cnt[n]+1) -> *out (block-reduced through LDS, one atomic per block)
__global__ __launch_bounds__(256)
void avglog_kernel(const float* __restrict__ cnt, float* __restrict__ out, int n)
{
  int i = blockIdx.x * 256 + threadIdx.x;
  float v = (i < n) ? logf(cnt[i] + 1.0f) : 0.0f;
  #pragma unroll
  for (int off = 16; off; off >>= 1) v += __shfl_xor(v, off, 32);
  __shared__ float ls[8];
  int lane = threadIdx.x & 31, wv = threadIdx.x >> 5;
  if (lane == 0) ls[wv] = v;
  __syncthreads();
  if (threadIdx.x == 0) {
    float s = 0.f;
    for (int w = 0; w < 8; ++w) s += ls[w];
    atomicAdd(out, s);
  }
}

// Build pna_post input (N,1664) bf16 = [xg | agg | agg*dlog/avg | agg*avg/dlog],
// agg = [mean, max, min, std]; launched over N*512 (one thread per agg element)
__global__ void pna_build_kernel(const float* __restrict__ xg, const float* __restrict__ sum,
                                 const float* __restrict__ sumsq, const unsigned* __restrict__ mx,
                                 const unsigned* __restrict__ mn, const float* __restrict__ cnt,
                                 const float* __restrict__ avgsum, uint16_t* __restrict__ out,
                                 long long n)
{
  long long i = (long long)blockIdx.x * blockDim.x + threadIdx.x;
  if (i >= n) return;
  int c = (int)(i % 512); long long node = i / 512;
  int grp = c >> 7, ch = c & 127;
  float cn = cnt[node];
  float denom = fmaxf(cn, 1.0f);
  size_t sc = (size_t)node * 128 + ch;
  float mean = sum[sc] / denom;
  float v;
  if (grp == 0)      v = mean;
  else if (grp == 1) v = (cn > 0.f) ? ord2f(mx[sc]) : 0.f;
  else if (grp == 2) v = (cn > 0.f) ? ord2f(mn[sc]) : 0.f;
  else { float m2 = sumsq[sc] / denom; v = sqrtf(fmaxf(m2 - mean * mean, 0.f) + 1e-5f); }
  float dlog = logf(denom + 1.0f);
  float avg = avgsum[0] * (1.0f / (float)N_NODES);
  size_t base = (size_t)node * 1664;
  out[base + 128 + c]  = f32_to_bf16(v);
  out[base + 640 + c]  = f32_to_bf16(v * (dlog / avg));
  out[base + 1152 + c] = f32_to_bf16(v * (avg / dlog));
  if (grp == 0) out[base + ch] = f32_to_bf16(xg[sc]);
}

// per-column sums for BatchNorm: one block per column
__global__ __launch_bounds__(256)
void colstat_kernel(const float* __restrict__ x, float* __restrict__ s1, float* __restrict__ s2,
                    int rows, int cols)
{
  int c = blockIdx.x;
  float a = 0.f, b = 0.f;
  for (int r = threadIdx.x; r < rows; r += 256) {
    float v = x[(size_t)r * cols + c];
    a += v; b += v * v;
  }
  #pragma unroll
  for (int off = 16; off; off >>= 1) { a += __shfl_xor(a, off, 32); b += __shfl_xor(b, off, 32); }
  __shared__ float la[8], lb[8];
  int lane = threadIdx.x & 31, wv = threadIdx.x >> 5;
  if (lane == 0) { la[wv] = a; lb[wv] = b; }
  __syncthreads();
  if (threadIdx.x == 0) {
    float A = 0.f, B2 = 0.f;
    for (int w = 0; w < 8; ++w) { A += la[w]; B2 += lb[w]; }
    s1[c] = A; s2[c] = B2;
  }
}

// x_gnn = (x_gnn + relu(batchnorm(p))) / 2
__global__ void bn_relu_mix_kernel(float* __restrict__ xg, const float* __restrict__ p,
                                   const float* __restrict__ s1, const float* __restrict__ s2,
                                   const float* __restrict__ bw, const float* __restrict__ bb,
                                   long long n)
{
  long long i = (long long)blockIdx.x * blockDim.x + threadIdx.x;
  if (i >= n) return;
  int c = (int)(i & 127);
  float mean = s1[c] * (1.0f / (float)N_NODES);
  float var  = s2[c] * (1.0f / (float)N_NODES) - mean * mean;
  float v = (p[i] - mean) * rsqrtf(var + 1e-5f) * bw[c] + bb[c];
  v = fmaxf(v, 0.f);
  xg[i] = (xg[i] + v) * 0.5f;
}

// out = ea + out/2   (eu2 output was written into `out`)
__global__ void eu_mix_kernel(const float* __restrict__ ea, float* __restrict__ out, long long n) {
  long long i = (long long)blockIdx.x * blockDim.x + threadIdx.x;
  if (i < n) out[i] = ea[i] + 0.5f * out[i];
}

// xc (B,320) f32 = [x_cls | xg[t0] | xg[t1]]
__global__ void xc_build_kernel(const float* __restrict__ xtab, const float* __restrict__ xg,
                                const int* __restrict__ t0, const int* __restrict__ t1,
                                float* __restrict__ xc, long long n)
{
  long long i = (long long)blockIdx.x * blockDim.x + threadIdx.x;
  if (i >= n) return;
  int c = (int)(i % 320); long long b = i / 320;
  float v;
  if (c < 64)        v = xtab[(size_t)(b * 7) * 64 + c];
  else if (c < 192)  v = xg[(size_t)t0[b] * 128 + (c - 64)];
  else               v = xg[(size_t)t1[b] * 128 + (c - 192)];
  xc[i] = v;
}

__global__ void avg2_kernel(float* __restrict__ a, const float* __restrict__ b, long long n) {
  long long i = (long long)blockIdx.x * blockDim.x + threadIdx.x;
  if (i < n) a[i] = (a[i] + b[i]) * 0.5f;
}

// x_cls' = (x_cls + xc[:, :64]) / 2
__global__ void cls_mix_kernel(const float* __restrict__ xtab, const float* __restrict__ xc,
                               float* __restrict__ xcls2, long long n)
{
  long long i = (long long)blockIdx.x * blockDim.x + threadIdx.x;
  if (i >= n) return;
  int c = (int)(i & 63); long long b = i >> 6;
  xcls2[i] = (xtab[(size_t)(b * 7) * 64 + c] + xc[(size_t)b * 320 + c]) * 0.5f;
}

// scatter emb = [xc[:,64:192] @ t0 ; xc[:,192:320] @ t1] into psum/pcnt
__global__ void pool_scatter_kernel(const float* __restrict__ xc, const int* __restrict__ t0,
                                    const int* __restrict__ t1, float* __restrict__ psum,
                                    float* __restrict__ pcnt, long long n)
{
  long long i = (long long)blockIdx.x * blockDim.x + threadIdx.x;
  if (i >= n) return;
  int c = (int)(i & 127); long long r = i >> 7;
  long long b; int off, idx;
  if (r < N_B) { b = r;        idx = t0[b]; off = 64;  }
  else         { b = r - N_B;  idx = t1[b]; off = 192; }
  atomicAdd(&psum[(size_t)idx * 128 + c], xc[(size_t)b * 320 + off + c]);
  if (c == 0) atomicAdd(&pcnt[idx], 1.0f);
}

__global__ void pool_mix_kernel(const float* __restrict__ xg, const float* __restrict__ psum,
                                const float* __restrict__ pcnt, float* __restrict__ out,
                                long long n)
{
  long long i = (long long)blockIdx.x * blockDim.x + threadIdx.x;
  if (i >= n) return;
  long long node = i >> 7;
  float cn = pcnt[node];
  float v = xg[i];
  if (cn > 0.f) v = (v + psum[i] / fmaxf(cn, 1.f)) * 0.5f;
  out[i] = v;
}

// ((x_tab' + tea)/2) flattened to (B,448) bf16; token0 = x_cls', tokens 1..6 from x_tab
__global__ void teaout_build_kernel(const float* __restrict__ xtab, const float* __restrict__ xcls2,
                                    const float* __restrict__ tea, uint16_t* __restrict__ out,
                                    long long n)
{
  long long i = (long long)blockIdx.x * blockDim.x + threadIdx.x;
  if (i >= n) return;
  int c = (int)(i % 448); long long b = i / 448;
  int t = c >> 6, ch = c & 63;
  float vt = (t == 0) ? xcls2[(size_t)b * 64 + ch] : xtab[(size_t)(b * 7 + t) * 64 + ch];
  out[i] = f32_to_bf16((vt + tea[i]) * 0.5f);
}

// ---------------------------------------------------------------------------
struct TfW {
  const uint16_t *qkv_w, *out_w, *ff1_w, *ff2_w;
  const float *qkv_b, *out_b, *ff1_b, *ff2_b, *n1w, *n1b, *n2w, *n2b;
};

extern "C" void kernel_launch(void* const* d_in, const int* in_sizes, int n_in,
                              void* d_out, int out_size, void* d_ws, size_t ws_size,
                              hipStream_t stream)
{
  (void)in_sizes; (void)n_in; (void)out_size; (void)ws_size;
  // Input order: top-level dict insertion order, params flattened as a JAX
  // pytree (dict keys sorted; 'L' < lowercase; within {'w','b'}: 'b' first):
  //  0 x  1 edge_index  2 edge_attr  3 target_edge_index  4 target_edge_attr
  //  5 L.bn.b       6 L.bn.w      7 L.eu1.b   8 L.eu1.w   9 L.eu2.b  10 L.eu2.w
  // 11 L.f1.b      12 L.f1.w     13 L.f2.b   14 L.f2.w   15 L.f3.b  16 L.f3.w
  // 17 L.fuse_ln.b 18 L.fuse_ln.w 19 L.fuse_norm.b 20 L.fuse_norm.w
  // 21 L.pna_edge_enc.b 22 L.pna_edge_enc.w 23 L.pna_lin.b 24 L.pna_lin.w
  // 25 L.pna_post.b 26 L.pna_post.w 27 L.pna_pre.b 28 L.pna_pre.w
  // 29..40 L.tab_conv {ff1.b,ff1.w,ff2.b,ff2.w,n1.b,n1.w,n2.b,n2.w,out.b,out.w,qkv.b,qkv.w}
  // 41 L.tab_norm.b 42 L.tab_norm.w 43 cls 44 edge_emb.b 45 edge_emb.w
  // 46 node_emb.b 47 node_emb.w 48..59 tab_conv {same order} 60 tab_norm.b 61 tab_norm.w
  auto F = [&](int i) { return (const float*)d_in[i]; };
  const int* edge_index = (const int*)d_in[1];
  const int* tgt_index  = (const int*)d_in[3];
  const int* e_src = edge_index;
  const int* e_dst = edge_index + N_EDGES;
  const int* t0 = tgt_index;
  const int* t1 = tgt_index + N_B;

  char* ws = (char*)d_ws;
  auto allocB = [&](size_t bytes) -> char* {
    char* p = ws; ws += (bytes + 255) & ~(size_t)255; return p;
  };
  auto allocF  = [&](size_t n) -> float*    { return (float*)allocB(n * 4); };
  auto allocU  = [&](size_t n) -> unsigned* { return (unsigned*)allocB(n * 4); };
  auto allocH  = [&](size_t n) -> uint16_t* { return (uint16_t*)allocB(n * 2); };
  auto castW = [&](int idx, size_t n) -> uint16_t* {
    uint16_t* d = allocH(n);
    cast_bf16_kernel<<<(int)((n + 255) / 256), 256, 0, stream>>>(F(idx), d, (int)n);
    return d;
  };

  // ---- weights -> bf16 (once per launch)
  uint16_t* w_qkv0  = castW(59, 192 * 64);
  uint16_t* w_out0  = castW(57, 64 * 64);
  uint16_t* w_ff1_0 = castW(49, 64 * 64);
  uint16_t* w_ff2_0 = castW(51, 64 * 64);
  uint16_t* w_qkvL  = castW(40, 192 * 64);
  uint16_t* w_outL  = castW(38, 64 * 64);
  uint16_t* w_ff1_L = castW(30, 64 * 64);
  uint16_t* w_ff2_L = castW(32, 64 * 64);
  uint16_t* w_eemb  = castW(45, 128 * 448);
  uint16_t* w_enc   = castW(22, 128 * 128);
  uint16_t* w_pre   = castW(28, 128 * 384);
  uint16_t* w_post  = castW(26, (size_t)128 * 1664);
  uint16_t* w_plin  = castW(24, 128 * 128);
  uint16_t* w_eu1   = castW(8, 128 * 384);
  uint16_t* w_eu2   = castW(10, 128 * 128);
  uint16_t* w_f1    = castW(12, (size_t)1280 * 320);
  uint16_t* w_f2    = castW(14, (size_t)1280 * 1280);
  uint16_t* w_f3    = castW(16, (size_t)320 * 1280);

  // ---- activations / scratch (f32 masters + bf16 mirrors where a GEMM consumes)
  float*    xg      = allocF((size_t)N_NODES * 128);
  float*    tea     = allocF((size_t)N_B * 448);
  uint16_t* teab    = allocH((size_t)N_B * 448);
  float*    xtab    = allocF((size_t)N_B * 448);
  uint16_t* xtabb   = allocH((size_t)N_B * 448);
  float*    eatok   = allocF((size_t)N_EDGES * 448);
  uint16_t* eatokb  = allocH((size_t)N_EDGES * 448);
  float*    s_qkv   = allocF((size_t)TF_CHUNK * 7 * 192);
  uint16_t* s_ob    = allocH((size_t)TF_CHUNK * 448);
  float*    s_t1    = allocF((size_t)TF_CHUNK * 448);
  uint16_t* s_t1b   = allocH((size_t)TF_CHUNK * 448);
  float*    s_t2    = allocF((size_t)TF_CHUNK * 448);
  float*    ea_emb  = allocF((size_t)N_EDGES * 128);
  uint16_t* ea_embb = allocH((size_t)N_EDGES * 128);
  float*    e_enc   = allocF((size_t)N_EDGES * 128);
  uint16_t* cat384b = allocH((size_t)N_EDGES * 384);
  float*    mbuf    = allocF((size_t)N_EDGES * 128);
  float*    segsum  = allocF((size_t)N_NODES * 128);
  float*    segsq   = allocF((size_t)N_NODES * 128);
  unsigned* segmx   = allocU((size_t)N_NODES * 128);
  unsigned* segmn   = allocU((size_t)N_NODES * 128);
  float*    segcnt  = allocF(N_NODES);
  float*    avgsum  = allocF(64);
  uint16_t* postinb = allocH((size_t)N_NODES * 1664);
  uint16_t* postoutb = allocH((size_t)N_NODES * 128);
  float*    pnaout  = allocF((size_t)N_NODES * 128);
  float*    bnsum   = allocF(128);
  float*    bnsq    = allocF(128);
  uint16_t* eutmpb  = allocH((size_t)N_EDGES * 128);
  float*    xc      = allocF((size_t)N_B * 320);
  float*    h0      = allocF((size_t)N_B * 320);
  uint16_t* h0b     = allocH((size_t)N_B * 320);
  uint16_t* h1b     = allocH((size_t)N_B * 1280);
  uint16_t* h2b     = allocH((size_t)N_B * 1280);
  float*    h3      = allocF((size_t)N_B * 320);
  float*    xcls2   = allocF((size_t)N_B * 64);
  float*    psum    = allocF((size_t)N_NODES * 128);
  float*    pcnt    = allocF(N_NODES);
  uint16_t* finb    = allocH((size_t)N_B * 448);

  float* out_xg = (float*)d_out;
  float* out_ea = out_xg + (size_t)N_NODES * 128;
  float* out_te = out_ea + (size_t)N_EDGES * 128;

  auto gemm = [&](const uint16_t* A, const uint16_t* W, const float* bias,
                  float* Cf, uint16_t* Cb, int M, int Nc, int K, int act) {
    dim3 g(Nc / 64, (M + 127) / 128);
    gemm_bf16<<<g, 256, 0, stream>>>(A, W, bias, Cf, Cb, M, Nc, K, act);
  };
  auto ew = [&](long long n) { return (int)((n + 255) / 256); };

  TfW tf0 { w_qkv0, w_out0, w_ff1_0, w_ff2_0,
            F(58), F(56), F(48), F(50), F(53), F(52), F(55), F(54) };
  TfW tfL { w_qkvL, w_outL, w_ff1_L, w_ff2_L,
            F(39), F(37), F(29), F(31), F(34), F(33), F(36), F(35) };

  auto tf_layer = [&](float* tok, uint16_t* tokb, int nseq, const TfW& w) {
    for (int cs = 0; cs < nseq; cs += TF_CHUNK) {
      int cn = nseq - cs; if (cn > TF_CHUNK) cn = TF_CHUNK;
      int T = cn * 7;
      float*    tk  = tok  + (size_t)cs * 448;
      uint16_t* tkb = tokb + (size_t)cs * 448;
      gemm(tkb, w.qkv_w, w.qkv_b, s_qkv, nullptr, T, 192, 64, 0);
      attn7_kernel<<<cn, 64, 0, stream>>>(s_qkv, s_ob);
      gemm(s_ob, w.out_w, w.out_b, s_t1, nullptr, T, 64, 64, 0);
      add_ln_kernel<<<(T + 7) / 8, 256, 0, stream>>>(tk, s_t1, w.n1w, w.n1b, tk, tkb, T, 64);
      gemm(tkb, w.ff1_w, w.ff1_b, nullptr, s_t1b, T, 64, 64, 1);
      gemm(s_t1b, w.ff2_w, w.ff2_b, s_t2, nullptr, T, 64, 64, 0);
      add_ln_kernel<<<(T + 7) / 8, 256, 0, stream>>>(tk, s_t2, w.n2w, w.n2b, tk, tkb, T, 64);
    }
  };

  // ---- zero atomic accumulators (required every call: graph replays reuse ws)
  hipMemsetAsync(segsum, 0, (size_t)N_NODES * 128 * 4, stream);
  hipMemsetAsync(segsq,  0, (size_t)N_NODES * 128 * 4, stream);
  hipMemsetAsync(segmx,  0x00, (size_t)N_NODES * 128 * 4, stream);   // 0 < ord(-inf) = 0x007FFFFF
  hipMemsetAsync(segmn,  0xFF, (size_t)N_NODES * 128 * 4, stream);   // > ord(+inf)
  hipMemsetAsync(segcnt, 0, (size_t)N_NODES * 4, stream);
  hipMemsetAsync(avgsum, 0, 4, stream);
  hipMemsetAsync(psum,   0, (size_t)N_NODES * 128 * 4, stream);
  hipMemsetAsync(pcnt,   0, (size_t)N_NODES * 4, stream);

  // ---- 1) node embedding
  node_emb_kernel<<<ew((long long)N_NODES * 128), 256, 0, stream>>>(
      F(0), F(47), F(46), xg, (long long)N_NODES * 128);

  // ---- 2) token buffers [cls | attr]
  build_tok_kernel<<<ew((long long)N_B * 448), 256, 0, stream>>>(
      F(43), F(4), tea, teab, (long long)N_B * 448);
  build_tok_kernel<<<ew((long long)N_EDGES * 448), 256, 0, stream>>>(
      F(43), F(2), eatok, eatokb, (long long)N_EDGES * 448);

  // ---- 3) top tab_conv + tab_norm on tea and ea
  tf_layer(tea, teab, N_B, tf0);
  add_ln_kernel<<<(N_B * 7 + 7) / 8, 256, 0, stream>>>(tea, nullptr, F(61), F(60),
                                                       tea, teab, N_B * 7, 64);
  tf_layer(eatok, eatokb, N_EDGES, tf0);
  add_ln_kernel<<<(N_EDGES * 7 + 7) / 8, 256, 0, stream>>>(eatok, nullptr, F(61), F(60),
                                                           eatok, eatokb, N_EDGES * 7, 64);

  // ---- 4) edge embedding: ea (E,448) -> (E,128)
  gemm(eatokb, w_eemb, F(44), ea_emb, ea_embb, N_EDGES, 128, 448, 0);

  // ---- 5) x_tab = LN(tf_L(tea))
  hipMemcpyAsync(xtab,  tea,  (size_t)N_B * 448 * 4, hipMemcpyDeviceToDevice, stream);
  hipMemcpyAsync(xtabb, teab, (size_t)N_B * 448 * 2, hipMemcpyDeviceToDevice, stream);
  tf_layer(xtab, xtabb, N_B, tfL);
  add_ln_kernel<<<(N_B * 7 + 7) / 8, 256, 0, stream>>>(xtab, nullptr, F(42), F(41),
                                                       xtab, nullptr, N_B * 7, 64);

  // ---- 6) PNA
  gemm(ea_embb, w_enc, F(21), e_enc, nullptr, N_EDGES, 128, 128, 0);
  cat3_kernel<<<ew((long long)N_EDGES * 384), 256, 0, stream>>>(
      xg, e_dst, e_src, e_enc, cat384b, (long long)N_EDGES * 384);
  gemm(cat384b, w_pre, F(27), mbuf, nullptr, N_EDGES, 128, 384, 0);
  pna_scatter_kernel<<<ew((long long)N_EDGES * 128), 256, 0, stream>>>(
      mbuf, e_dst, segsum, segsq, segmx, segmn, (long long)N_EDGES * 128);
  count_kernel<<<(N_EDGES + 255) / 256, 256, 0, stream>>>(e_dst, segcnt, N_EDGES);
  avglog_kernel<<<(N_NODES + 255) / 256, 256, 0, stream>>>(segcnt, avgsum, N_NODES);
  pna_build_kernel<<<ew((long long)N_NODES * 512), 256, 0, stream>>>(
      xg, segsum, segsq, segmx, segmn, segcnt, avgsum, postinb, (long long)N_NODES * 512);
  gemm(postinb, w_post, F(25), nullptr, postoutb, N_NODES, 128, 1664, 0);
  gemm(postoutb, w_plin, F(23), pnaout, nullptr, N_NODES, 128, 128, 0);
  colstat_kernel<<<128, 256, 0, stream>>>(pnaout, bnsum, bnsq, N_NODES, 128);
  bn_relu_mix_kernel<<<ew((long long)N_NODES * 128), 256, 0, stream>>>(
      xg, pnaout, bnsum, bnsq, F(6), F(5), (long long)N_NODES * 128);

  // ---- 7) edge update -> output ea
  cat3_kernel<<<ew((long long)N_EDGES * 384), 256, 0, stream>>>(
      xg, e_src, e_dst, ea_emb, cat384b, (long long)N_EDGES * 384);
  gemm(cat384b, w_eu1, F(7), nullptr, eutmpb, N_EDGES, 128, 384, 1);
  gemm(eutmpb, w_eu2, F(9), out_ea, nullptr, N_EDGES, 128, 128, 0);
  eu_mix_kernel<<<ew((long long)N_EDGES * 128), 256, 0, stream>>>(
      ea_emb, out_ea, (long long)N_EDGES * 128);

  // ---- 8) fuse MLP
  xc_build_kernel<<<ew((long long)N_B * 320), 256, 0, stream>>>(
      xtab, xg, t0, t1, xc, (long long)N_B * 320);
  add_ln_kernel<<<(N_B + 7) / 8, 256, 0, stream>>>(xc, nullptr, F(18), F(17), h0, h0b, N_B, 320);
  gemm(h0b, w_f1, F(11), nullptr, h1b, N_B, 1280, 320, 2);
  gemm(h1b, w_f2, F(13), nullptr, h2b, N_B, 1280, 1280, 2);
  gemm(h2b, w_f3, F(15), h3, nullptr, N_B, 320, 1280, 0);
  add_ln_kernel<<<(N_B + 7) / 8, 256, 0, stream>>>(h3, nullptr, F(20), F(19), h0, nullptr, N_B, 320);
  avg2_kernel<<<ew((long long)N_B * 320), 256, 0, stream>>>(xc, h0, (long long)N_B * 320);
  cls_mix_kernel<<<ew((long long)N_B * 64), 256, 0, stream>>>(
      xtab, xc, xcls2, (long long)N_B * 64);

  // ---- 9) pooling -> output x_gnn
  pool_scatter_kernel<<<ew((long long)2 * N_B * 128), 256, 0, stream>>>(
      xc, t0, t1, psum, pcnt, (long long)2 * N_B * 128);
  pool_mix_kernel<<<ew((long long)N_NODES * 128), 256, 0, stream>>>(
      xg, psum, pcnt, out_xg, (long long)N_NODES * 128);

  // ---- 10) tea_out = edge_emb(((x_tab' + tea)/2).reshape(B,448))
  teaout_build_kernel<<<ew((long long)N_B * 448), 256, 0, stream>>>(
      xtab, xcls2, tea, finb, (long long)N_B * 448);
  gemm(finb, w_eemb, F(44), out_te, nullptr, N_B, 128, 448, 0);
}
